// TGMTransformerDecoder_28089086116359
// MI455X (gfx1250) — matrix operand order repaired and checked by reading, not compiled, not run on hardware
//
#include <hip/hip_runtime.h>

typedef __attribute__((ext_vector_type(16))) _Float16 v16h;
typedef __attribute__((ext_vector_type(8)))  _Float16 v8h;
typedef __attribute__((ext_vector_type(8)))  float    v8f;

#define BB    2
#define CC    256
#define NHW   16      // cluster grid 16x16
#define HW    128     // H = W
#define VHW   64      // VH = VW
#define NSP   256
#define PP    4096    // VH*VW
#define TAU_INV (1.0f/0.07f)
#define EPSV  1e-16f

// CDNA5 async global->LDS copy path (guarded: falls back to plain stores if
// the builtin is absent on this toolchain). The builtin takes int* operands
// (global addrspace(1) source, LDS addrspace(3) destination).
#if defined(__gfx1250__) && __has_builtin(__builtin_amdgcn_global_load_async_to_lds_b32)
#define HAVE_ASYNC_LDS 1
typedef __attribute__((address_space(1))) int Gas32;
typedef __attribute__((address_space(3))) int Las32;
__device__ __forceinline__ void wait_asynccnt0() {
#if __has_builtin(__builtin_amdgcn_s_wait_asynccnt)
  __builtin_amdgcn_s_wait_asynccnt(0);
#else
  asm volatile("s_wait_asynccnt 0" ::: "memory");
#endif
}
#endif

// ---------------------------------------------------------------------------
// Kernel 0: zero fill
// ---------------------------------------------------------------------------
__global__ void zero_f32(float* __restrict__ p, int n) {
  int i = blockIdx.x * blockDim.x + threadIdx.x;
  if (i < n) p[i] = 0.0f;
}

// ---------------------------------------------------------------------------
// Kernel 1: per-cell cosine-affinity GEMM via v_wmma_f32_16x16x32_f16,
// + mask + /TAU, write aff output, 2x2 downsample, 9-way softmax (att_c),
// atomic scatter of att_c into sums[b][idx].
// One workgroup (128 thr = 4 waves) per (batch, 8x8-pixel cell).
// ---------------------------------------------------------------------------
__launch_bounds__(128)
__global__ void aff_wmma_kernel(const float* __restrict__ qf,   // (B,NSP,C)
                                const float* __restrict__ kf,   // (B,C,H,W)
                                const int*   __restrict__ spix, // (B,9,VH,VW)
                                float* __restrict__ aff_out,    // (B,9,H,W)
                                float* __restrict__ att_buf,    // (B,9,P)
                                float* __restrict__ sums)       // (B,NSP+1)
{
  __shared__ __align__(32) float    qraw[9][CC];   // raw neighbor q
  __shared__ __align__(32) _Float16 bq[16][CC];    // normalized q, f16 (B operand, N-major)
  __shared__ __align__(32) _Float16 ks[64][CC];    // pixel k, f16 (A operand, pixel-major)
  __shared__ float qrn[16];                        // 1/max(||q||,1e-12)
  __shared__ float rkn[64];                        // 1/max(||k||,1e-12)
  __shared__ float affs[9][64];                    // masked affinity / TAU
  __shared__ float es[9][16];                      // exp(aff_down)

  const int tid  = threadIdx.x;
  const int blk  = blockIdx.x;
  const int b    = blk >> 8;
  const int cell = blk & 255;
  const int gy   = cell >> 4, gx = cell & 15;
  const int y0   = gy * 8,    x0 = gx * 8;

  // --- stage 3x3 neighbor q vectors (zero-padded outside grid) ---
  // In-range dwords go through the CDNA5 async global->LDS data path
  // (ASYNCcnt-tracked, no VGPR round trip); OOB lanes store zeros via DS.
  for (int e = tid; e < 9 * CC; e += 128) {
    int n = e >> 8, c = e & 255;
    int cy = gy + n / 3 - 1, cx = gx + n % 3 - 1;
    if (cy >= 0 && cy < NHW && cx >= 0 && cx < NHW) {
      const float* src = &qf[((size_t)b * NSP + cy * NHW + cx) * CC + c];
#if defined(HAVE_ASYNC_LDS)
      __builtin_amdgcn_global_load_async_to_lds_b32(
          (Gas32*)(const void*)src, (Las32*)(void*)&qraw[n][c], 0, 0);
#else
      qraw[n][c] = *src;
#endif
    } else {
      qraw[n][c] = 0.0f;
    }
  }
#if defined(HAVE_ASYNC_LDS)
  wait_asynccnt0();
#endif
  __syncthreads();

  // --- q norms ---
  if (tid < 16) {
    float r = 0.0f;
    if (tid < 9) {
      float s = 0.0f;
      for (int c = 0; c < CC; ++c) { float v = qraw[tid][c]; s += v * v; }
      r = 1.0f / fmaxf(sqrtf(s), 1e-12f);
    }
    qrn[tid] = r;
  }
  __syncthreads();

  // --- normalized q -> f16 B operand (rows 9..15 zero) ---
  for (int e = tid; e < 16 * CC; e += 128) {
    int n = e >> 8, c = e & 255;
    float v = (n < 9) ? qraw[n][c] * qrn[n] : 0.0f;
    bq[n][c] = (_Float16)v;
  }

  // --- stage 8x8 k patch, all 256 channels, as f16 A operand ---
  const float* kb = kf + (size_t)b * CC * HW * HW;
  for (int it = tid; it < CC * 8; it += 128) {
    int c = it >> 3, row = it & 7;
    const float* src = kb + ((size_t)c * HW + (y0 + row)) * HW + x0;
    float4 v0 = *(const float4*)(src);
    float4 v1 = *(const float4*)(src + 4);
    int p = row * 8;
    ks[p + 0][c] = (_Float16)v0.x; ks[p + 1][c] = (_Float16)v0.y;
    ks[p + 2][c] = (_Float16)v0.z; ks[p + 3][c] = (_Float16)v0.w;
    ks[p + 4][c] = (_Float16)v1.x; ks[p + 5][c] = (_Float16)v1.y;
    ks[p + 6][c] = (_Float16)v1.z; ks[p + 7][c] = (_Float16)v1.w;
  }
  __syncthreads();

  // --- per-pixel k norms ---
  if (tid < 64) {
    float s = 0.0f;
    for (int c = 0; c < CC; ++c) { float v = (float)ks[tid][c]; s += v * v; }
    rkn[tid] = 1.0f / fmaxf(sqrtf(s), 1e-12f);
  }
  __syncthreads();

  // --- GEMM: wave w owns pixels [16w,16w+16) x 16 neighbor slots, K=256 ---
  const int w = tid >> 5, lane = tid & 31;
  const int hi = lane >> 4, ln = lane & 15;
  v8f acc = {};
  const _Float16* arow = &ks[16 * w + ln][0];   // A row M = lane%16
  const _Float16* brow = &bq[ln][0];            // B col N = lane%16, K contiguous
  for (int kk = 0; kk < 8; ++kk) {
    union { v16h v; v8h h[2]; } ua;
    ua.h[0] = *(const v8h*)(arow + 32 * kk + 8 * hi);        // K = 32kk+8hi .. +7
    ua.h[1] = *(const v8h*)(arow + 32 * kk + 16 + 8 * hi);   // K = 32kk+16+8hi .. +7
    union { v16h v; v8h h[2]; } ub;
    ub.h[0] = *(const v8h*)(brow + 32 * kk + 8 * hi);        // K = 32kk+8hi .. +7
    ub.h[1] = *(const v8h*)(brow + 32 * kk + 16 + 8 * hi);   // K = 32kk+16+8hi .. +7
    v16h bv = ub.v;
    acc = __builtin_amdgcn_wmma_f32_16x16x32_f16(
        false, ua.v, false, bv, (short)0, acc, false, false);
  }

  // --- epilogue: D VGPR r -> row M = r + 8*hi, col N = lane%16 ---
  for (int r = 0; r < 8; ++r) {
    int p = 16 * w + r + 8 * hi;      // pixel within cell
    int n = ln;                        // neighbor slot
    if (n < 9) {
      float cosv = acc[r] * rkn[p] * qrn[n];
      int ky = n / 3, kx = n % 3;
      bool masked = (ky == 0 && gy == 0) || (ky == 2 && gy == NHW - 1) ||
                    (kx == 0 && gx == 0) || (kx == 2 && gx == NHW - 1);
      float val = masked ? -__builtin_inff() : cosv * TAU_INV;
      int y = y0 + (p >> 3), x = x0 + (p & 7);
      aff_out[(((size_t)b * 9 + n) * HW + y) * HW + x] = val;
      affs[n][p] = val;
    }
  }
  __syncthreads();

  // --- 2x2 downsample + exp ---
  for (int e = tid; e < 144; e += 128) {
    int n = e % 9, dpos = e / 9;
    int dy = dpos >> 2, dx = dpos & 3;
    int p0 = (2 * dy) * 8 + 2 * dx;
    float m4 = 0.25f * (affs[n][p0] + affs[n][p0 + 1] +
                        affs[n][p0 + 8] + affs[n][p0 + 9]);
    es[n][dpos] = expf(m4);                    // exp(-inf) == 0 for masked
  }
  __syncthreads();

  // --- softmax over 9, store att_c, atomic-accumulate sums ---
  for (int e = tid; e < 144; e += 128) {
    int n = e % 9, dpos = e / 9;
    int dy = dpos >> 2, dx = dpos & 3;
    float s = 0.0f;
    for (int j = 0; j < 9; ++j) s += es[j][dpos];
    float att = es[n][dpos] / (s + EPSV);
    int y2 = gy * 4 + dy, x2 = gx * 4 + dx;
    int pd = y2 * VHW + x2;
    att_buf[((size_t)b * 9 + n) * PP + pd] = att;
    int idx = spix[(((size_t)b * 9 + n) * VHW + y2) * VHW + x2];
    if (idx < NSP) atomicAdd(&sums[b * (NSP + 1) + idx], att);
  }
}

// ---------------------------------------------------------------------------
// Kernel 2: att_p = att_c / (sums[b, idx(batch0)] + eps); scatter att_p * v
// and att_p * k_down into out_cls / out_query with global f32 atomics.
// One block per (b, n, y2 row); threads coalesce over x2.
// ---------------------------------------------------------------------------
__launch_bounds__(256)
__global__ void scatter_kernel(const float* __restrict__ kf,      // (B,C,H,W)
                               const float* __restrict__ vf,      // (B,C,VH,VW)
                               const int*   __restrict__ spix,    // (B,9,VH,VW)
                               const float* __restrict__ att_buf, // (B,9,P)
                               const float* __restrict__ sums,    // (B,NSP+1)
                               float* __restrict__ out_query,     // (B,NSP,C)
                               float* __restrict__ out_cls)       // (B,NSP,C)
{
  __shared__ float attp[VHW];
  __shared__ int   idxs[VHW];

  int blk = blockIdx.x;                 // B*9*VHW
  int b   = blk / (9 * VHW);
  int rem = blk % (9 * VHW);
  int n   = rem / VHW, y2 = rem % VHW;
  int t   = threadIdx.x;

  if (t < VHW) {
    int x2 = t;
    int pd = y2 * VHW + x2;
    float att  = att_buf[((size_t)b * 9 + n) * PP + pd];
    // reference quirk: denom gathered with batch-0 indices for ALL batches
    int idx0   = spix[((size_t)n * VHW + y2) * VHW + x2];
    float den  = sums[b * (NSP + 1) + idx0];
    attp[t] = att / (den + EPSV);
    idxs[t] = spix[(((size_t)b * 9 + n) * VHW + y2) * VHW + x2];
  }
  __syncthreads();

  int x2 = t & 63;
  int c0 = t >> 6;                       // 0..3 channel phase
  int idx   = idxs[x2];
  float wgt = attp[x2];
  if (idx >= NSP || wgt == 0.0f) return; // bin NSP is dropped; zero adds no-op

  const float* kb = kf + (size_t)b * CC * HW * HW;
  const float* vb = vf + (size_t)b * CC * VHW * VHW;
  float* oq = out_query + ((size_t)b * NSP + idx) * CC;
  float* oc = out_cls   + ((size_t)b * NSP + idx) * CC;

  for (int c = c0; c < CC; c += 4) {
    const float* kr = kb + ((size_t)c * HW + 2 * y2) * HW + 2 * x2;
    if (c + 4 < CC)
      __builtin_prefetch(kb + ((size_t)(c + 4) * HW + 2 * y2) * HW + 2 * x2, 0, 1);
    float kd = 0.25f * (kr[0] + kr[1] + kr[HW] + kr[HW + 1]);   // down2x(k)
    float vv = vb[((size_t)c * VHW + y2) * VHW + x2];
    atomicAdd(&oq[c], wgt * kd);
    atomicAdd(&oc[c], wgt * vv);
  }
}

// ---------------------------------------------------------------------------
extern "C" void kernel_launch(void* const* d_in, const int* in_sizes, int n_in,
                              void* d_out, int out_size, void* d_ws, size_t ws_size,
                              hipStream_t stream) {
  const float* qf   = (const float*)d_in[0];   // (2,256,256)
  const float* kf   = (const float*)d_in[1];   // (2,256,128,128)
  const float* vf   = (const float*)d_in[2];   // (2,256,64,64)
  const int*   spix = (const int*)  d_in[3];   // (2,9,64,64)

  float* out       = (float*)d_out;
  float* out_query = out;                      // 131072
  float* out_cls   = out + BB * NSP * CC;      // +131072
  float* aff_out   = out + 2 * BB * NSP * CC;  // +262144

  float* ws      = (float*)d_ws;
  float* sums    = ws;                         // B*(NSP+1) = 514, padded to 1024
  float* att_buf = ws + 1024;                  // B*9*P = 73728

  // zero scatter destinations + sums (aff/att_buf are fully overwritten)
  zero_f32<<<(2 * BB * NSP * CC + 255) / 256, 256, 0, stream>>>(out, 2 * BB * NSP * CC);
  zero_f32<<<4, 256, 0, stream>>>(sums, 1024);

  aff_wmma_kernel<<<BB * NHW * NHW, 128, 0, stream>>>(qf, kf, spix, aff_out, att_buf, sums);
  scatter_kernel<<<BB * 9 * VHW, 256, 0, stream>>>(kf, vf, spix, att_buf, sums,
                                                   out_query, out_cls);
}
